// MyMultiHeadAttention_80436147519654
// MI455X (gfx1250) — compile-verified, matrix operand order checked by reading, not run on hardware
//
#include <hip/hip_runtime.h>
#include <hip/hip_bf16.h>

typedef unsigned short u16;
typedef unsigned int   u32;
typedef __attribute__((ext_vector_type(16))) __bf16 v16bf;
typedef __attribute__((ext_vector_type(8)))  float  v8f;
typedef __attribute__((ext_vector_type(4)))  u32    u32x4;
typedef __attribute__((ext_vector_type(8)))  int    i32x8;
typedef __attribute__((ext_vector_type(4)))  int    i32x4;

struct BFPair { uint4 lo, hi; };

static __device__ __forceinline__ u16 f2bfu(float f) {
    unsigned u = __builtin_bit_cast(unsigned, f);
    unsigned r = u + 0x7FFFu + ((u >> 16) & 1u);   // round-to-nearest-even
    return (u16)(r >> 16);
}

// Load a 16-element bf16 WMMA fragment: two 16-byte chunks (global or LDS).
static __device__ __forceinline__ v16bf ldfrag(const u16* p, int o0, int o1) {
    BFPair t;
    t.lo = *reinterpret_cast<const uint4*>(p + o0);
    t.hi = *reinterpret_cast<const uint4*>(p + o1);
    return __builtin_bit_cast(v16bf, t);
}

static __device__ __forceinline__ v8f wmma_bf16(v16bf a, v16bf b, v8f c) {
    return __builtin_amdgcn_wmma_f32_16x16x32_bf16(false, a, false, b, (short)0, c, false, false);
}

static __device__ __forceinline__ float redmax16(float v) {
    v = fmaxf(v, __shfl_xor(v, 1, 32));
    v = fmaxf(v, __shfl_xor(v, 2, 32));
    v = fmaxf(v, __shfl_xor(v, 4, 32));
    v = fmaxf(v, __shfl_xor(v, 8, 32));
    return v;
}
static __device__ __forceinline__ float redsum16(float v) {
    v += __shfl_xor(v, 1, 32);
    v += __shfl_xor(v, 2, 32);
    v += __shfl_xor(v, 4, 32);
    v += __shfl_xor(v, 8, 32);
    return v;
}

// ---------------------------------------------------------------------------
// TDM: 1-D contiguous tile (nelem bf16 elements) global -> LDS.
// D# per CDNA5 ISA ch.8: group0 = {count=1, lds_addr, global_addr, type=2},
// group1 = {data_size=2B, tensor_dim0=nelem, tensor_dim1=1, tile_dim0=nelem}.
// Groups 2/3 zero (tensor <= 2D). Tracked by TENSORcnt.
// This toolchain exposes the 6-arg builtin (extra int32x8 group, then cpol).
// ---------------------------------------------------------------------------
static __device__ __forceinline__ void tdm_load_1d(u32 lds_addr, const u16* gsrc, u32 nelem) {
    unsigned long long ga = (unsigned long long)(uintptr_t)gsrc;
    u32x4 g0;
    g0[0] = 1u;                                          // count = 1 valid descriptor
    g0[1] = lds_addr;                                    // LDS byte address
    g0[2] = (u32)ga;                                     // global_addr[31:0]
    g0[3] = (u32)((ga >> 32) & 0x1FFFFFFu) | (2u << 30); // global_addr[56:32] | type=2
    i32x8 g1;
    g1[0] = (int)(1u << 16);                             // data_size = 1 (2 bytes)
    g1[1] = (int)((nelem & 0xFFFFu) << 16);              // tensor_dim0[15:0]
    g1[2] = (int)(((nelem >> 16) & 0xFFFFu) | (1u << 16)); // tensor_dim0[31:16] | tensor_dim1=1
    g1[3] = (int)((nelem & 0xFFFFu) << 16);              // tile_dim0 = nelem
    g1[4] = 0;                                           // tile_dim1/2 unused
    g1[5] = (int)nelem;                                  // tensor_dim0_stride (unused, sane)
    g1[6] = 0;
    g1[7] = 0;
    i32x4 z4 = {0, 0, 0, 0};
    i32x8 z8 = {0, 0, 0, 0, 0, 0, 0, 0};
    __builtin_amdgcn_tensor_load_to_lds(g0, g1, z4, z4, z8, 0);
}

// ---------------------------------------------------------------------------
// Kernel 0: f32 -> bf16 conversion (grid-stride)
// ---------------------------------------------------------------------------
__global__ void cvt_bf16_kernel(const float* __restrict__ src, u16* __restrict__ dst, int n) {
    for (int i = blockIdx.x * blockDim.x + threadIdx.x; i < n; i += gridDim.x * blockDim.x)
        dst[i] = f2bfu(src[i]);
}

// ---------------------------------------------------------------------------
// Kernel 1: QKV projection, NT GEMM, 32x64 register tile per wave
//           (2 A-frags x 4 B-frags -> 8 WMMAs per K-step) + RoPE / V-transpose.
// grid (128 m-tiles, 2, 3 projections), block 256 (8 waves; wave -> 64-col group)
// ---------------------------------------------------------------------------
__global__ __launch_bounds__(256)
void qkv_rope_kernel(const u16* __restrict__ xh,
                     const u16* __restrict__ wqh, const u16* __restrict__ wkh,
                     const u16* __restrict__ wvh,
                     const int* __restrict__ position,
                     u16* __restrict__ qh, u16* __restrict__ kh, u16* __restrict__ vth) {
    const int wave = threadIdx.x >> 5;
    const int lane = threadIdx.x & 31;
    const int mt   = blockIdx.x;                 // 32-row tile
    const int h    = blockIdx.y * 8 + wave;      // 64-col group == head index
    const int proj = blockIdx.z;

    const u16* W = (proj == 0) ? wqh : (proj == 1) ? wkh : wvh;

    const int i  = lane & 15;
    const int kg = lane >> 4;

    const u16* arow[2];
    arow[0] = xh + (size_t)(mt * 32 + i) * 1024;
    arow[1] = xh + (size_t)(mt * 32 + 16 + i) * 1024;
    const u16* brow[4];
    #pragma unroll
    for (int ns = 0; ns < 4; ++ns)
        brow[ns] = W + (size_t)(h * 64 + ns * 16 + i) * 1024;

    v8f acc[2][4] = {{{}, {}, {}, {}}, {{}, {}, {}, {}}};

    for (int kb = 0; kb < 1024; kb += 32) {
        __builtin_prefetch(arow[0] + kb + 256, 0, 3);
        __builtin_prefetch(arow[1] + kb + 256, 0, 3);

        const int o0 = kb + kg * 8;
        const int o1 = kb + 16 + kg * 8;
        v16bf a0 = ldfrag(arow[0], o0, o1);
        v16bf a1 = ldfrag(arow[1], o0, o1);
        v16bf b0 = ldfrag(brow[0], o0, o1);
        v16bf b1 = ldfrag(brow[1], o0, o1);
        v16bf b2 = ldfrag(brow[2], o0, o1);
        v16bf b3 = ldfrag(brow[3], o0, o1);
        acc[0][0] = wmma_bf16(a0, b0, acc[0][0]);
        acc[0][1] = wmma_bf16(a0, b1, acc[0][1]);
        acc[0][2] = wmma_bf16(a0, b2, acc[0][2]);
        acc[0][3] = wmma_bf16(a0, b3, acc[0][3]);
        acc[1][0] = wmma_bf16(a1, b0, acc[1][0]);
        acc[1][1] = wmma_bf16(a1, b1, acc[1][1]);
        acc[1][2] = wmma_bf16(a1, b2, acc[1][2]);
        acc[1][3] = wmma_bf16(a1, b3, acc[1][3]);
    }

    // Epilogue. D-frag element (m = r + 8*kg, n = i); dh = ns*16 + i.
    if (proj < 2) {
        u16* dst = (proj == 0) ? qh : kh;
        float invf[4];
        #pragma unroll
        for (int ns = 0; ns < 4; ++ns)
            invf[ns] = powf(10000.0f, -(float)((ns * 16 + i) & ~1) / 64.0f);

        #pragma unroll
        for (int ms = 0; ms < 2; ++ms) {
            #pragma unroll
            for (int r = 0; r < 8; ++r) {
                const int t  = mt * 32 + ms * 16 + r + 8 * kg;  // token over B*L
                const int bb = t >> 11;
                const int l  = t & 2047;
                const float pos = (float)position[t];
                u16* drow = dst + ((size_t)(bb * 16 + h) * 2048 + l) * 64;
                #pragma unroll
                for (int ns = 0; ns < 4; ++ns) {
                    const int dh = ns * 16 + i;
                    float s, c;
                    sincosf(pos * invf[ns], &s, &c);
                    const float v = acc[ms][ns][r];
                    const float partner = __shfl_xor(v, 1, 32);  // dh^1 lives in lane^1
                    const float res = (dh & 1) ? (partner * s + v * c)
                                               : (v * c - partner * s);
                    drow[dh] = f2bfu(res);
                }
            }
        }
    } else {
        // V stored transposed per (b,h): Vt[Dh][L]
        #pragma unroll
        for (int ms = 0; ms < 2; ++ms) {
            #pragma unroll
            for (int r = 0; r < 8; ++r) {
                const int t  = mt * 32 + ms * 16 + r + 8 * kg;
                const int bb = t >> 11;
                const int l  = t & 2047;
                #pragma unroll
                for (int ns = 0; ns < 4; ++ns) {
                    const int dh = ns * 16 + i;
                    vth[((size_t)(bb * 16 + h) * 64 + dh) * 2048 + l] = f2bfu(acc[ms][ns][r]);
                }
            }
        }
    }
}

// ---------------------------------------------------------------------------
// Kernel 2: causal flash attention, one wave per (b, h, 16-query tile).
// K tiles staged into LDS by the Tensor Data Mover (TENSOR_LOAD_TO_LDS,
// tracked by TENSORcnt); P relayout via LDS (DScnt). V via clause'd b128.
// grid (32, 16, 2), block 128 (4 waves)
// ---------------------------------------------------------------------------
__global__ __launch_bounds__(128)
void flash_attn_kernel(const u16* __restrict__ qh, const u16* __restrict__ kh,
                       const u16* __restrict__ vth, u16* __restrict__ oh) {
    __shared__ __align__(64) u16 Ktile[4][32][64];  // per-wave K tile (TDM destination)
    __shared__ __align__(64) u16 Pbuf[4][16][32];   // per-wave P tile (D-frag -> A-frag)

    const int wave = threadIdx.x >> 5;
    const int lane = threadIdx.x & 31;
    const int qt = blockIdx.x * 4 + wave;
    const int h  = blockIdx.y;
    const int b  = blockIdx.z;

    const size_t bh = (size_t)(b * 16 + h);
    const u16* Qb = qh  + bh * 2048 * 64;
    const u16* Kb = kh  + bh * 2048 * 64;
    const u16* Vb = vth + bh * 64 * 2048;

    const int i  = lane & 15;
    const int kg = lane >> 4;

    const u32 kt_lds = (u32)(uintptr_t)&Ktile[wave][0][0];

    // Q A-fragments for Dh=64 (two K=32 chunks), resident in registers
    const u16* qrow = Qb + (size_t)(qt * 16 + i) * 64;
    const v16bf qf0 = ldfrag(qrow, kg * 8,      16 + kg * 8);
    const v16bf qf1 = ldfrag(qrow, 32 + kg * 8, 48 + kg * 8);

    v8f oacc[4] = {{}, {}, {}, {}};
    float mrow[8], lrow[8];
    #pragma unroll
    for (int r = 0; r < 8; ++r) { mrow[r] = -1e30f; lrow[r] = 0.0f; }

    const int nblocks = (qt * 16 + 15) / 32 + 1;
    for (int kk = 0; kk < nblocks; ++kk) {
        const int kbase = kk * 32;

        // ---- DMA the 32x64 K tile (rows kbase..kbase+31) into LDS ----
        // DS reads of Ktile/Pbuf are unordered vs. TENSOR ops: drain DScnt first.
        asm volatile("s_wait_dscnt 0" ::: "memory");
        tdm_load_1d(kt_lds, Kb + (size_t)kbase * 64, 32 * 64);
        asm volatile("s_wait_tensorcnt 0" ::: "memory");

        // ---- scores: S[16 x 32] = Q (16x64) . K^T, 4 WMMAs (K frags from LDS) ----
        const u16* krow_a = &Ktile[wave][i][0];
        const u16* krow_b = &Ktile[wave][16 + i][0];
        v8f sa = {}, sb = {};
        sa = wmma_bf16(qf0, ldfrag(krow_a, kg * 8,      16 + kg * 8), sa);
        sa = wmma_bf16(qf1, ldfrag(krow_a, 32 + kg * 8, 48 + kg * 8), sa);
        sb = wmma_bf16(qf0, ldfrag(krow_b, kg * 8,      16 + kg * 8), sb);
        sb = wmma_bf16(qf1, ldfrag(krow_b, 32 + kg * 8, 48 + kg * 8), sb);

        const bool domask = (kbase + 31) > (qt * 16);
        const int key_a = kbase + i;
        const int key_b = kbase + 16 + i;

        // ---- online softmax over the 32-key block ----
        float pa[8], pb[8];
        #pragma unroll
        for (int r = 0; r < 8; ++r) {
            const int q = qt * 16 + r + 8 * kg;
            float va = sa[r] * 0.125f;           // 1/sqrt(64)
            float vb = sb[r] * 0.125f;
            if (domask) {
                if (key_a > q) va = -1e9f;
                if (key_b > q) vb = -1e9f;
            }
            const float rmax = redmax16(fmaxf(va, vb));
            const float mnew = fmaxf(mrow[r], rmax);
            const float alpha = __expf(mrow[r] - mnew);
            pa[r] = __expf(va - mnew);
            pb[r] = __expf(vb - mnew);
            const float rs = redsum16(pa[r] + pb[r]);
            lrow[r] = lrow[r] * alpha + rs;
            mrow[r] = mnew;
            oacc[0][r] *= alpha; oacc[1][r] *= alpha;
            oacc[2][r] *= alpha; oacc[3][r] *= alpha;
        }

        // ---- relayout P: D-fragment -> A-fragment via LDS ----
        #pragma unroll
        for (int r = 0; r < 8; ++r) {
            const int m = r + 8 * kg;
            Pbuf[wave][m][i]      = f2bfu(pa[r]);
            Pbuf[wave][m][16 + i] = f2bfu(pb[r]);
        }
        asm volatile("s_wait_dscnt 0" ::: "memory");   // wave-local LDS visibility

        const v16bf pf = ldfrag(&Pbuf[wave][i][0], kg * 8, 16 + kg * 8);

        // ---- O[16x64] += P (16x32) . V (32x64): 4 WMMAs (Vt rows contiguous) ----
        #pragma unroll
        for (int ntile = 0; ntile < 4; ++ntile) {
            const u16* vrow = Vb + (size_t)(ntile * 16 + i) * 2048 + kbase;
            oacc[ntile] = wmma_bf16(pf, ldfrag(vrow, kg * 8, 16 + kg * 8), oacc[ntile]);
        }
    }

    // ---- normalize and store to [B, L, H*Dh] (bf16 feeds the output projection) ----
    #pragma unroll
    for (int r = 0; r < 8; ++r) {
        const float inv = 1.0f / lrow[r];
        const int tok = b * 2048 + qt * 16 + r + 8 * kg;
        u16* orow = oh + (size_t)tok * 1024 + h * 64;
        #pragma unroll
        for (int ntile = 0; ntile < 4; ++ntile)
            orow[ntile * 16 + i] = f2bfu(oacc[ntile][r] * inv);
    }
}

// ---------------------------------------------------------------------------
// Kernel 3: output projection, NT GEMM, 32x64 register tile per wave, f32 out
// grid (128, 2), block 256
// ---------------------------------------------------------------------------
__global__ __launch_bounds__(256)
void out_proj_kernel(const u16* __restrict__ oh, const u16* __restrict__ woh,
                     float* __restrict__ out) {
    const int wave = threadIdx.x >> 5;
    const int lane = threadIdx.x & 31;
    const int mt  = blockIdx.x;
    const int ng  = blockIdx.y * 8 + wave;   // 64-col group

    const int i  = lane & 15;
    const int kg = lane >> 4;

    const u16* arow[2];
    arow[0] = oh + (size_t)(mt * 32 + i) * 1024;
    arow[1] = oh + (size_t)(mt * 32 + 16 + i) * 1024;
    const u16* brow[4];
    #pragma unroll
    for (int ns = 0; ns < 4; ++ns)
        brow[ns] = woh + (size_t)(ng * 64 + ns * 16 + i) * 1024;

    v8f acc[2][4] = {{{}, {}, {}, {}}, {{}, {}, {}, {}}};

    for (int kb = 0; kb < 1024; kb += 32) {
        __builtin_prefetch(arow[0] + kb + 256, 0, 3);
        __builtin_prefetch(arow[1] + kb + 256, 0, 3);

        const int o0 = kb + kg * 8;
        const int o1 = kb + 16 + kg * 8;
        v16bf a0 = ldfrag(arow[0], o0, o1);
        v16bf a1 = ldfrag(arow[1], o0, o1);
        v16bf b0 = ldfrag(brow[0], o0, o1);
        v16bf b1 = ldfrag(brow[1], o0, o1);
        v16bf b2 = ldfrag(brow[2], o0, o1);
        v16bf b3 = ldfrag(brow[3], o0, o1);
        acc[0][0] = wmma_bf16(a0, b0, acc[0][0]);
        acc[0][1] = wmma_bf16(a0, b1, acc[0][1]);
        acc[0][2] = wmma_bf16(a0, b2, acc[0][2]);
        acc[0][3] = wmma_bf16(a0, b3, acc[0][3]);
        acc[1][0] = wmma_bf16(a1, b0, acc[1][0]);
        acc[1][1] = wmma_bf16(a1, b1, acc[1][1]);
        acc[1][2] = wmma_bf16(a1, b2, acc[1][2]);
        acc[1][3] = wmma_bf16(a1, b3, acc[1][3]);
    }

    #pragma unroll
    for (int ms = 0; ms < 2; ++ms) {
        #pragma unroll
        for (int r = 0; r < 8; ++r) {
            const int row = mt * 32 + ms * 16 + r + 8 * kg;
            float* orow = out + (size_t)row * 1024 + ng * 64;
            #pragma unroll
            for (int ns = 0; ns < 4; ++ns)
                orow[ns * 16 + i] = acc[ms][ns][r];
        }
    }
}

// ---------------------------------------------------------------------------
// Host launcher
// Workspace layout (bytes), total 48 MiB:
//   xh   @ 0         (B*L*D bf16, 8 MiB)
//   wqh  @ 8388608   (2 MiB)   wkh @ 10485760   wvh @ 12582912   woh @ 14680064
//   qh   @ 16777216  (8 MiB)   kh  @ 25165824   vth @ 33554432   oh @ 41943040
// ---------------------------------------------------------------------------
extern "C" void kernel_launch(void* const* d_in, const int* in_sizes, int n_in,
                              void* d_out, int out_size, void* d_ws, size_t ws_size,
                              hipStream_t stream) {
    const float* x        = (const float*)d_in[0];
    const int*   position = (const int*)d_in[1];
    const float* Wq       = (const float*)d_in[2];
    const float* Wk       = (const float*)d_in[3];
    const float* Wv       = (const float*)d_in[4];
    const float* Wo       = (const float*)d_in[5];
    float* out = (float*)d_out;

    char* ws = (char*)d_ws;
    u16* xh  = (u16*)(ws + 0);
    u16* wqh = (u16*)(ws + 8388608);
    u16* wkh = (u16*)(ws + 10485760);
    u16* wvh = (u16*)(ws + 12582912);
    u16* woh = (u16*)(ws + 14680064);
    u16* qh  = (u16*)(ws + 16777216);
    u16* kh  = (u16*)(ws + 25165824);
    u16* vth = (u16*)(ws + 33554432);
    u16* oh  = (u16*)(ws + 41943040);

    const int NX = 2 * 2048 * 1024;   // 4,194,304
    const int NW = 1024 * 1024;       // 1,048,576

    cvt_bf16_kernel<<<4096, 256, 0, stream>>>(x,  xh,  NX);
    cvt_bf16_kernel<<<2048, 256, 0, stream>>>(Wq, wqh, NW);
    cvt_bf16_kernel<<<2048, 256, 0, stream>>>(Wk, wkh, NW);
    cvt_bf16_kernel<<<2048, 256, 0, stream>>>(Wv, wvh, NW);
    cvt_bf16_kernel<<<2048, 256, 0, stream>>>(Wo, woh, NW);

    qkv_rope_kernel<<<dim3(128, 2, 3), 256, 0, stream>>>(xh, wqh, wkh, wvh, position,
                                                         qh, kh, vth);
    flash_attn_kernel<<<dim3(32, 16, 2), 128, 0, stream>>>(qh, kh, vth, oh);
    out_proj_kernel<<<dim3(128, 2), 256, 0, stream>>>(oh, woh, out);
}
